// MFDenseLayer_20667382629121
// MI455X (gfx1250) — matrix operand order repaired and checked by reading, not compiled, not run on hardware
//
#include <hip/hip_runtime.h>
#include <hip/hip_bf16.h>

// Problem constants (reference: B,N,K,IN,OUT = 64,10000,16,64,64)
constexpr int BB   = 64;
constexpr int NN   = 10000;
constexpr int KK   = 16;
constexpr int INF  = 64;
constexpr int OUTF = 64;
constexpr int NPW  = 16;              // nodes per workgroup -> grid = 625
constexpr int PITCH = 72;             // ushort pitch for 64-wide rows: 144B = 9*16B (aligned, bank-friendly)
constexpr int PLANE = 64 * PITCH;     // one 64x64 bf16 tile in LDS (ushorts)

typedef __attribute__((ext_vector_type(16))) __bf16 v16bf;
typedef __attribute__((ext_vector_type(8)))  float  v8f;

union FragAB { uint4 q[2]; v16bf v; };
union BF2    { unsigned int u; __hip_bfloat162 h; };

__device__ __forceinline__ unsigned short f2bf(float f) {
    unsigned int u = __float_as_uint(f);                 // round-to-nearest-even bf16
    return (unsigned short)((u + 0x7FFFu + ((u >> 16) & 1u)) >> 16);
}
__device__ __forceinline__ unsigned int packbf2(float lo, float hi) {
    return (unsigned int)f2bf(lo) | ((unsigned int)f2bf(hi) << 16);
}

__global__ __launch_bounds__(256)
void mf_dense_kernel(const float* __restrict__ x,  const float* __restrict__ w1,
                     const float* __restrict__ w2, const float* __restrict__ b1,
                     const float* __restrict__ b2, float* __restrict__ out)
{
    // LDS: w2 transposed bf16 [k][o][i] (144KB) + 2 node weight tiles + 2 x tiles + bias
    __shared__ unsigned short s_w2t[KK * PLANE];
    __shared__ unsigned short s_wnt[2 * PLANE];
    __shared__ unsigned short s_x  [2 * PLANE];
    __shared__ float          s_bias[2 * OUTF];

    const int tid = threadIdx.x;

    // ---- Stage w2 (fp32 [k][i*64+o]) -> LDS bf16 transposed s_w2t[k][o][i] (once per WG)
    for (int qi = tid; qi < (KK * INF * OUTF) / 4; qi += 256) {
        const int k  = qi >> 10;          // 1024 float4 per k-plane
        const int r  = qi & 1023;
        const int i  = r >> 4;            // IN index
        const int o4 = (r & 15) << 2;     // OUT base (4 consecutive o)
        float4 f = reinterpret_cast<const float4*>(w2)[qi];
        unsigned short* dst = &s_w2t[k * PLANE + i];
        dst[(o4 + 0) * PITCH] = f2bf(f.x);
        dst[(o4 + 1) * PITCH] = f2bf(f.y);
        dst[(o4 + 2) * PITCH] = f2bf(f.z);
        dst[(o4 + 3) * PITCH] = f2bf(f.w);
    }
    __syncthreads();

    for (int it = 0; it < NPW / 2; ++it) {
        const int n0 = blockIdx.x * NPW + it * 2;   // two nodes per step

        // ---- Stage activations x[b, n, :] -> bf16 s_x[node][b][i]
        {
            const int nodeS = tid >> 7;             // 0/1
            const int r     = tid & 127;
            const int brow  = r >> 1;               // batch row 0..63
            const int half  = (r & 1) * 32;         // 32 floats each
            const int n     = n0 + nodeS;
            const float* src = x + ((size_t)brow * NN + n) * INF + half;
            unsigned int* dst = reinterpret_cast<unsigned int*>(
                &s_x[nodeS * PLANE + brow * PITCH + half]);
            #pragma unroll
            for (int v = 0; v < 8; ++v) {
                float4 f = reinterpret_cast<const float4*>(src)[v];
                dst[2 * v + 0] = packbf2(f.x, f.y);
                dst[2 * v + 1] = packbf2(f.z, f.w);
            }
            if (it + 1 < NPW / 2)                   // warm next node-pair (global_prefetch_b8)
                __builtin_prefetch(src + 2 * INF, 0, 1);
        }

        // ---- Per-node bias[n,:] = b1[n,:] @ b2  (tiny, L2-resident operands)
        if (tid < 2 * OUTF) {
            const int nd = tid >> 6;
            const int o  = tid & 63;
            const int n  = n0 + nd;
            float s = 0.f;
            #pragma unroll
            for (int k = 0; k < KK; ++k)
                s = fmaf(b1[n * KK + k], b2[k * OUTF + o], s);
            s_bias[nd * OUTF + o] = s;
        }

        // ---- Synthesize WnT[o][i] = sum_k w1[n,k] * w2T[k][o][i]  (packed bf16 FMA)
        {
            const int nodeS = tid >> 7;
            const int rem   = tid & 127;
            const int obase = rem >> 3;             // 0..15
            const int i0    = (rem & 7) << 3;       // 0,8,...,56
            const int n     = n0 + nodeS;
            __hip_bfloat162 w1s[KK];
            #pragma unroll
            for (int k = 0; k < KK; ++k) {          // scalar (wave-uniform) loads, splat to bf16x2
                __hip_bfloat16 hb = __float2bfloat16(w1[n * KK + k]);
                w1s[k].x = hb; w1s[k].y = hb;
            }
            #pragma unroll
            for (int qq = 0; qq < 4; ++qq) {
                const int o = obase + 16 * qq;
                BF2 a0, a1, a2, a3;
                a0.u = 0u; a1.u = 0u; a2.u = 0u; a3.u = 0u;
                #pragma unroll
                for (int k = 0; k < KK; ++k) {
                    const uint4 wq = *reinterpret_cast<const uint4*>(
                        &s_w2t[k * PLANE + o * PITCH + i0]);
                    BF2 t;
                    t.u = wq.x; a0.h = __hfma2(w1s[k], t.h, a0.h);
                    t.u = wq.y; a1.h = __hfma2(w1s[k], t.h, a1.h);
                    t.u = wq.z; a2.h = __hfma2(w1s[k], t.h, a2.h);
                    t.u = wq.w; a3.h = __hfma2(w1s[k], t.h, a3.h);
                }
                uint4 st; st.x = a0.u; st.y = a1.u; st.z = a2.u; st.w = a3.u;
                *reinterpret_cast<uint4*>(&s_wnt[nodeS * PLANE + o * PITCH + i0]) = st;
            }
        }
        __syncthreads();

        // ---- WMMA: waves 0-3 -> node n0 (rows 0-63), waves 4-7 -> node n0+1
        {
            const int wv     = tid >> 5;
            const int lane   = tid & 31;
            const int nd     = wv >> 2;
            const int node   = n0 + nd;
            const int mbase  = (wv & 3) * 16;
            const int laneLo = lane & 15;
            const int hiHalf = lane >> 4;           // 0 or 1
            const unsigned short* Xb = &s_x  [nd * PLANE];
            const unsigned short* Wb = &s_wnt[nd * PLANE];

            // A (16x32 bf16): lane row = mbase+laneLo; K split {0-7,16-23}/{8-15,24-31}
            FragAB fa0, fa1;
            {
                const int base = (mbase + laneLo) * PITCH + hiHalf * 8;
                fa0.q[0] = *reinterpret_cast<const uint4*>(&Xb[base +  0]);
                fa0.q[1] = *reinterpret_cast<const uint4*>(&Xb[base + 16]);
                fa1.q[0] = *reinterpret_cast<const uint4*>(&Xb[base + 32]);
                fa1.q[1] = *reinterpret_cast<const uint4*>(&Xb[base + 48]);
            }
            #pragma unroll
            for (int nt = 0; nt < 4; ++nt) {
                const int o = nt * 16 + laneLo;
                // B (32x16 bf16): lane column = o (row of WnT); K split {0-7,16-23}/{8-15,24-31}
                FragAB fb0, fb1;
                const int bbase = o * PITCH + hiHalf * 8;
                fb0.q[0] = *reinterpret_cast<const uint4*>(&Wb[bbase +  0]);
                fb0.q[1] = *reinterpret_cast<const uint4*>(&Wb[bbase + 16]);
                fb1.q[0] = *reinterpret_cast<const uint4*>(&Wb[bbase + 32]);
                fb1.q[1] = *reinterpret_cast<const uint4*>(&Wb[bbase + 48]);

                const float bv = s_bias[nd * OUTF + o];
                v8f c = {bv, bv, bv, bv, bv, bv, bv, bv};
                c = __builtin_amdgcn_wmma_f32_16x16x32_bf16(false, fa0.v, false, fb0.v,
                                                            (short)0, c, false, false);
                c = __builtin_amdgcn_wmma_f32_16x16x32_bf16(false, fa1.v, false, fb1.v,
                                                            (short)0, c, false, false);
                // C/D layout: VGPR j -> rows mbase+j (lanes 0-15) / mbase+j+8 (lanes 16-31)
                #pragma unroll
                for (int j = 0; j < 8; ++j) {
                    const int rowj = mbase + j + hiHalf * 8;
                    out[((size_t)rowj * NN + node) * OUTF + o] = c[j];
                }
            }
        }
        __syncthreads();
    }
}

extern "C" void kernel_launch(void* const* d_in, const int* in_sizes, int n_in,
                              void* d_out, int out_size, void* d_ws, size_t ws_size,
                              hipStream_t stream) {
    const float* x  = (const float*)d_in[0];
    const float* w1 = (const float*)d_in[1];
    const float* w2 = (const float*)d_in[2];
    const float* b1 = (const float*)d_in[3];
    const float* b2 = (const float*)d_in[4];
    float* out = (float*)d_out;
    (void)in_sizes; (void)n_in; (void)out_size; (void)d_ws; (void)ws_size;

    dim3 grid(NN / NPW);   // 625
    dim3 block(256);       // 8 wave32 waves
    hipLaunchKernelGGL(mf_dense_kernel, grid, block, 0, stream, x, w1, w2, b1, b2, out);
}
